// NearestEmbedding_41120016892003
// MI455X (gfx1250) — matrix operand order repaired and checked by reading, not compiled, not run on hardware
//
#include <hip/hip_runtime.h>
#include <hip/hip_bf16.h>

// NearestEmbedding: BatchNorm1d(x) -> argmin_k ||xn - w_k||^2 -> gather w.
// Distances via score_k = 0.5*||w_k||^2 - xn.w_k  (same argmin).
// GEMM via v_wmma_f32_16x16x32_f16 (wave32); codebook staged to LDS with
// global_load_async_to_lds_b128 (ASYNCcnt) overlapped with x normalization.
// k-loop software-pipelined so prev-tile argmin VALU fills the WMMA->VALU
// hazard window (no v_nops); lane argmin reduction via ds_swizzle xor steps.

typedef __attribute__((ext_vector_type(16))) _Float16 v16h;
typedef __attribute__((ext_vector_type(8)))  float    v8f;

__device__ __forceinline__ unsigned pack2h(float a, float b) {
    union { _Float16 h[2]; unsigned u; } p;
    p.h[0] = (_Float16)a; p.h[1] = (_Float16)b;
    return p.u;
}

// ---------------- kernel 1: zero the reduction accumulators -------------
__global__ __launch_bounds__(128) void ne_zero(float* acc) {
    acc[threadIdx.x] = 0.0f;   // 128 floats: sums[64] + sumsq[64]
}

// ---------------- kernel 2: column sums / sums-of-squares ---------------
__global__ __launch_bounds__(256) void ne_reduce(const float* __restrict__ x,
                                                 float* __restrict__ sums,
                                                 float* __restrict__ sumsq,
                                                 int N) {
    const int tid = threadIdx.x;
    const int c = tid & 63;        // column
    const int g = tid >> 6;        // row group 0..3
    float s = 0.0f, s2 = 0.0f;
    for (int row = blockIdx.x * 4 + g; row < N; row += gridDim.x * 4) {
        float v = x[(size_t)row * 64 + c];
        s += v; s2 += v * v;
    }
    __shared__ float ls[256], ls2[256];
    ls[tid] = s; ls2[tid] = s2;
    __syncthreads();
    if (g == 0) {
        s  = ls[c]  + ls[c + 64]  + ls[c + 128]  + ls[c + 192];
        s2 = ls2[c] + ls2[c + 64] + ls2[c + 128] + ls2[c + 192];
        atomicAdd(&sums[c], s);
        atomicAdd(&sumsq[c], s2);
    }
}

// ------- kernel 3: scale/shift, 0.5*||w||^2, fp32 -> f16 weight pack ----
__global__ __launch_bounds__(512) void ne_finalize(const float* __restrict__ sums,
                                                   const float* __restrict__ sumsq,
                                                   const float* __restrict__ gamma,
                                                   const float* __restrict__ beta,
                                                   const float* __restrict__ weight,
                                                   float* __restrict__ scale,
                                                   float* __restrict__ shift,
                                                   float* __restrict__ w2h,
                                                   unsigned* __restrict__ whq,
                                                   float invN) {
    const int tid = threadIdx.x;          // 0..511 = codeword id
    if (tid < 64) {
        float mean = sums[tid] * invN;
        float var  = sumsq[tid] * invN - mean * mean;
        float sc   = gamma[tid] * rsqrtf(var + 1e-5f);
        scale[tid] = sc;
        shift[tid] = beta[tid] - mean * sc;
    }
    const float* wr = weight + (size_t)tid * 64;
    unsigned*    wq = whq    + (size_t)tid * 32;
    float acc = 0.0f;
#pragma unroll
    for (int d = 0; d < 64; d += 2) {
        float a = wr[d], b = wr[d + 1];
        acc += a * a + b * b;
        wq[d >> 1] = pack2h(a, b);
    }
    w2h[tid] = 0.5f * acc;
}

// ---------------- kernel 4: WMMA distance + argmin + gather -------------
// Block: 256 threads = 8 waves; each wave owns 16 rows -> 128 rows/block.
__global__ __launch_bounds__(256) void ne_main(const float* __restrict__ x,
                                               const float* __restrict__ weight,
                                               const float* __restrict__ scale,
                                               const float* __restrict__ shift,
                                               const float* __restrict__ w2h,
                                               const unsigned* __restrict__ whq,
                                               float* __restrict__ out) {
    __shared__ __align__(16) float    sScale[64];
    __shared__ __align__(16) float    sShift[64];
    __shared__ __align__(16) float    sW2[512];
    __shared__ __align__(16) unsigned sX[8][16 * 32];   // per-wave 16x64 halves (packed 2/uint)
    __shared__ __align__(16) unsigned sWh[512 * 32];    // full f16 codebook: 512 x 64 halves (64 KB)
    __shared__ int sIdx[128];

    const int tid  = threadIdx.x;
    const int wave = tid >> 5;
    const int lane = tid & 31;
    const unsigned m   = lane & 15;     // A row / B,C column within 16
    const unsigned hiK = lane >> 4;     // lane half selector

    // ---- kick off async copy of the whole f16 codebook into LDS ----
    // Low 32 bits of a flat LDS pointer are the LDS byte offset (aperture spec).
    {
        const unsigned ldsBase = (unsigned)(size_t)(&sWh[0]);
#pragma unroll
        for (int it = 0; it < 16; ++it) {
            int e = it * 256 + tid;                       // uint4 index 0..4095
            unsigned lds = ldsBase + (unsigned)e * 16u;
            unsigned long long ga =
                (unsigned long long)(const void*)(whq + (size_t)e * 4);
            asm volatile("global_load_async_to_lds_b128 %0, %1, off"
                         :: "v"(lds), "v"(ga) : "memory");
        }
    }

    // ---- stage small constants into LDS (overlaps with async DMA) ----
    if (tid < 64) { sScale[tid] = scale[tid]; sShift[tid] = shift[tid]; }
    sW2[tid]       = w2h[tid];
    sW2[tid + 256] = w2h[tid + 256];
    __syncthreads();

    // ---- normalize this wave's 16x64 x tile into LDS as packed halves ----
    const int row0 = blockIdx.x * 128 + wave * 16;
    const float4* X4 = reinterpret_cast<const float4*>(x);
    const float4* SC = reinterpret_cast<const float4*>(sScale);
    const float4* SH = reinterpret_cast<const float4*>(sShift);
    unsigned* sXw = &sX[wave][0];
#pragma unroll
    for (int it = 0; it < 8; ++it) {
        int e  = it * 32 + lane;            // 0..255 float4 elements of the tile
        int r  = e >> 4;                    // row 0..15
        int c4 = e & 15;                    // float4 column 0..15
        float4 xv = X4[(size_t)(row0 + r) * 16 + c4];
        float4 sc = SC[c4];
        float4 sh = SH[c4];
        float y0 = xv.x * sc.x + sh.x;
        float y1 = xv.y * sc.y + sh.y;
        float y2 = xv.z * sc.z + sh.z;
        float y3 = xv.w * sc.w + sh.w;
        sXw[r * 32 + c4 * 2]     = pack2h(y0, y1);
        sXw[r * 32 + c4 * 2 + 1] = pack2h(y2, y3);
    }

    // ---- build the two A fragments (K = 0..31 and 32..63) in contiguous k order:
    // lane L (m=L&15, hiK=L>>4), v16h slot i: v=i>>1, h=i&1,
    // k = 16*hiK + (v<<1) + h  -> packed uint holds (k, k+1).
    union AFrag { v16h v; unsigned u[8]; } a0, a1;
#pragma unroll
    for (int v = 0; v < 8; ++v) {
        unsigned dp = 8u * hiK + (unsigned)v;
        a0.u[v] = sXw[m * 32 + dp];
        a1.u[v] = sXw[m * 32 + 16 + dp];
    }

    // ---- running argmin state: C layout => lane covers rows M=j+8*hiK, col N=lane&15
    float best[8];
    int   bidx[8];
#pragma unroll
    for (int j = 0; j < 8; ++j) { best[j] = 3.4e38f; bidx[j] = 0x7fffffff; }

    // ---- codebook DMA must have landed before we read sWh ----
    asm volatile("s_wait_asynccnt 0" ::: "memory");
    __syncthreads();

    // ---- software-pipelined k-loop: WMMA for tile kt, argmin for tile kt-1 ----
    const uint4* S4 = reinterpret_cast<const uint4*>(sWh);
    union BFrag { v16h v; uint4 q[2]; } b0, b1;

    // prologue: tile 0
    int   kprev = (int)m;
    {
        const int base4 = kprev * 8 + 2 * (int)hiK;
        b0.q[0] = S4[base4];     b0.q[1] = S4[base4 + 1];
        b1.q[0] = S4[base4 + 4]; b1.q[1] = S4[base4 + 5];
    }
    float w2prev = sW2[kprev];
    v8f cprev = {};
    cprev = __builtin_amdgcn_wmma_f32_16x16x32_f16(false, a0.v, false, b0.v,
                                                   (short)0, cprev, false, false);
    cprev = __builtin_amdgcn_wmma_f32_16x16x32_f16(false, a1.v, false, b1.v,
                                                   (short)0, cprev, false, false);

    for (int kt = 1; kt < 32; ++kt) {
        const int kcur = kt * 16 + (int)m;
        const int base4 = kcur * 8 + 2 * (int)hiK;
        b0.q[0] = S4[base4];     b0.q[1] = S4[base4 + 1];
        b1.q[0] = S4[base4 + 4]; b1.q[1] = S4[base4 + 5];
        const float w2cur = sW2[kcur];

        v8f ccur = {};
        ccur = __builtin_amdgcn_wmma_f32_16x16x32_f16(false, a0.v, false, b0.v,
                                                      (short)0, ccur, false, false);
        ccur = __builtin_amdgcn_wmma_f32_16x16x32_f16(false, a1.v, false, b1.v,
                                                      (short)0, ccur, false, false);

        // process previous tile (independent of in-flight WMMAs -> fills hazard)
#pragma unroll
        for (int j = 0; j < 8; ++j) {
            float score = w2prev - cprev[j];
            if (score < best[j]) { best[j] = score; bidx[j] = kprev; }
        }
        cprev = ccur; w2prev = w2cur; kprev = kcur;
    }
    // epilogue: last tile
#pragma unroll
    for (int j = 0; j < 8; ++j) {
        float score = w2prev - cprev[j];
        if (score < best[j]) { best[j] = score; bidx[j] = kprev; }
    }

    // ---- argmin across the 16 lanes of each half (cols), ties -> lower index ----
    // ds_swizzle group-of-32 xor mode: imm = and(0x1f) | (xor << 10)
#define NE_RSTEP(IMM)                                                          \
    {                                                                          \
        float os = __int_as_float(                                             \
            __builtin_amdgcn_ds_swizzle(__float_as_int(best[j]), (IMM)));      \
        int oi = __builtin_amdgcn_ds_swizzle(bidx[j], (IMM));                  \
        if (os < best[j] || (os == best[j] && oi < bidx[j])) {                 \
            best[j] = os; bidx[j] = oi;                                        \
        }                                                                      \
    }
#pragma unroll
    for (int j = 0; j < 8; ++j) {
        NE_RSTEP(0x041f)   // xor 1
        NE_RSTEP(0x081f)   // xor 2
        NE_RSTEP(0x101f)   // xor 4
        NE_RSTEP(0x201f)   // xor 8
    }
#undef NE_RSTEP

    if ((lane & 15) == 0) {
#pragma unroll
        for (int j = 0; j < 8; ++j)
            sIdx[wave * 16 + (int)hiK * 8 + j] = bidx[j];
    }
    __syncthreads();

    // ---- gather exact fp32 codebook rows, coalesced float4 stores ----
    const float4* W4 = reinterpret_cast<const float4*>(weight);
    float4*       O4 = reinterpret_cast<float4*>(out);
    const size_t  obase = (size_t)blockIdx.x * 128 * 16;
#pragma unroll
    for (int it = 0; it < 8; ++it) {
        int e  = it * 256 + tid;          // 0..2047
        int r  = e >> 4;                  // block-local row 0..127
        int c4 = e & 15;
        int idx = sIdx[r];
        O4[obase + (size_t)r * 16 + c4] = W4[(size_t)idx * 16 + c4];
    }
}

extern "C" void kernel_launch(void* const* d_in, const int* in_sizes, int n_in,
                              void* d_out, int out_size, void* d_ws, size_t ws_size,
                              hipStream_t stream) {
    const float* x      = (const float*)d_in[0];
    const float* weight = (const float*)d_in[1];
    const float* gamma  = (const float*)d_in[2];
    const float* beta   = (const float*)d_in[3];
    float* out = (float*)d_out;

    const int D = 64, K = 512;
    const int N = in_sizes[0] / D;        // 131072

    // workspace layout (floats): sums[64] sumsq[64] scale[64] shift[64] w2h[512] | f16 weights
    float* ws     = (float*)d_ws;
    float* sums   = ws;
    float* sumsq  = ws + 64;
    float* scale  = ws + 128;
    float* shift  = ws + 192;
    float* w2h    = ws + 256;
    unsigned* whq = (unsigned*)(ws + 768);   // K*D/2 = 16384 uints (64 KB)

    ne_zero<<<1, 128, 0, stream>>>(sums);
    ne_reduce<<<256, 256, 0, stream>>>(x, sums, sumsq, N);
    ne_finalize<<<1, K, 0, stream>>>(sums, sumsq, gamma, beta, weight,
                                     scale, shift, w2h, whq, 1.0f / (float)N);
    ne_main<<<N / 128, 256, 0, stream>>>(x, weight, scale, shift, w2h, whq, out);
}
